// Classification3Stage_13975823582045
// MI455X (gfx1250) — matrix operand order repaired and checked by reading, not compiled, not run on hardware
//
#include <hip/hip_runtime.h>

// ===========================================================================
// 3-stage routed classifier, MI455X (gfx1250, wave32, WMMA f32_16x16x32_f16).
//
// Pipeline (all on `stream`):
//   1) NCHW fp32 -> [N,128] f16 compaction (one HBM pass; afterwards x stays
//      L2-resident: 10.5MB << 192MB L2).
//   2) Weight prep: cm weights [E][Cin][Cout] -> [E][Cout][Cin] f16 so B
//      fragments are contiguous-K 16B loads; conv weights are already
//      [Cout][Cin], just converted.
//   3) Coarse MLP (128->32->32->16) with WMMA, per-16-pixel wave tiles,
//      argmax -> inds1, CE loss 0.
//   4) MoE grouped-GEMM for the index-conditioned stages: bucket pixels by
//      expert (hist + padded scan + scatter), one wave per 16-row tile:
//        - route stage2 (experts=inds1)  -> inds12
//        - route stage3 (experts=inds12) -> inds123 -> d_out
//        - loss stage2 (buckets by i1_gt, offsets -1/0/+1 share A frags,
//          fully unrolled with private LDS tiles -> 36 interleavable WMMAs)
//        - loss stage3 (buckets by i12_gt)
//   5) finalize: losses[i] = (ce_sum/N) * (mask_cnt/N).
// ===========================================================================

typedef __attribute__((ext_vector_type(16))) _Float16 v16h;
typedef __attribute__((ext_vector_type(8)))  _Float16 v8h;
typedef __attribute__((ext_vector_type(8)))  float    v8f;

#define DEVINL static __device__ __forceinline__

constexpr int NPIX = 40960;   // 1 * 160 * 256

// ---------------------------------------------------------------- utilities
DEVINL void lds_fence() {
#if __has_builtin(__builtin_amdgcn_s_wait_dscnt)
  __builtin_amdgcn_s_wait_dscnt(0);
  asm volatile("" ::: "memory");
#else
  asm volatile("s_wait_dscnt 0" ::: "memory");
#endif
}

DEVINL int clampi(int v, int lo, int hi) { return v < lo ? lo : (v > hi ? hi : v); }

DEVINL v8f wmma_f16(v16h a, v16h b, v8f c) {
  // D(16x16,f32) = A(16x32,f16) x B(32x16,f16) + C
  return __builtin_amdgcn_wmma_f32_16x16x32_f16(false, a, false, b, (short)0, c,
                                                false, false);
}

// B fragment of W^T (32x16 tile). Weight storage: [ncols][K] f16, K contiguous.
// ISA layout: lane (N=m,hh) holds K = {k0+8hh .. +7} U {k0+16+8hh .. +7}
// -> two 16B loads per lane.
DEVINL v16h load_bfrag(const _Float16* wt, int Kdim, int ncol, int k0, int m, int hh) {
  const _Float16* p = wt + (size_t)(ncol + m) * Kdim + k0 + hh * 8;
  v8h lo = *(const v8h*)p;
  v8h hi = *(const v8h*)(p + 16);
  v16h b;
#pragma unroll
  for (int i = 0; i < 8; ++i) { b[i] = lo[i]; b[8 + i] = hi[i]; }
  return b;
}

// A fragment (16x32). Lane (m,hh) holds row M=m,
// K = {k0+8hh .. +7} U {k0+16+8hh .. +7} -> two 16B loads from the row.
DEVINL v16h load_afrag_row(const _Float16* row, int k0, int hh) {
  v16h a;
  if (row) {
    const _Float16* p = row + k0 + hh * 8;
    v8h s0 = *(const v8h*)p;
    v8h s1 = *(const v8h*)(p + 16);
#pragma unroll
    for (int i = 0; i < 8; ++i) { a[i] = s0[i]; a[8 + i] = s1[i]; }
  } else {
#pragma unroll
    for (int i = 0; i < 16; ++i) a[i] = (_Float16)0.0f;
  }
  return a;
}

DEVINL v16h load_afrag_lds(const _Float16* lds, int m, int hh) {
  const _Float16* p = lds + m * 32 + hh * 8;
  v8h s0 = *(const v8h*)p;
  v8h s1 = *(const v8h*)(p + 16);
  v16h a;
#pragma unroll
  for (int i = 0; i < 8; ++i) { a[i] = s0[i]; a[8 + i] = s1[i]; }
  return a;
}

// bias + leaky-relu on C fragments, spill f16 to LDS tile [16][32] so the next
// layer can reload in A order (cross-lane C->A transpose).
DEVINL void bias_lrelu_to_lds(v8f c0, v8f c1, const float* b, _Float16* lds,
                              int m, int hh) {
  lds_fence();  // WAR vs. previous layer's A loads from this tile
  float b0 = b[m], b1v = b[m + 16];
#pragma unroll
  for (int r = 0; r < 8; ++r) {
    float v0 = c0[r] + b0;  v0 = v0 > 0.f ? v0 : 0.01f * v0;
    float v1 = c1[r] + b1v; v1 = v1 > 0.f ? v1 : 0.01f * v1;
    int row = r + 8 * hh;
    lds[row * 32 + m]      = (_Float16)v0;
    lds[row * 32 + 16 + m] = (_Float16)v1;
  }
  lds_fence();  // RAW before A reload (same wave; hardware in-order, fence for compiler)
}

// 16-lane (half-wave) reductions; xor masks < 16 stay within the half.
DEVINL float rmax16(float v) {
#pragma unroll
  for (int s = 8; s; s >>= 1) v = fmaxf(v, __shfl_xor(v, s, 32));
  return v;
}
DEVINL float rsum16(float v) {
#pragma unroll
  for (int s = 8; s; s >>= 1) v += __shfl_xor(v, s, 32);
  return v;
}
DEVINL void rargmax16(float& v, int& i) {  // first-occurrence (lowest idx) ties
#pragma unroll
  for (int s = 8; s; s >>= 1) {
    float ov = __shfl_xor(v, s, 32);
    int   oi = __shfl_xor(i, s, 32);
    if (ov > v || (ov == v && oi < i)) { v = ov; i = oi; }
  }
}

// 128 -> 32 -> 32 -> 32 expert MLP on one 16-row tile. 12 WMMAs.
DEVINL void mlp_128_32_32_32(const v16h xa[4],
                             const _Float16* w1t, const float* b1,
                             const _Float16* w2t, const float* b2,
                             const _Float16* w3t, const float* b3,
                             _Float16* lds, int m, int hh, v8f& o0, v8f& o1) {
  v8f c0 = {}, c1 = {};
#pragma unroll
  for (int kc = 0; kc < 4; ++kc) {
    c0 = wmma_f16(xa[kc], load_bfrag(w1t, 128, 0,  kc * 32, m, hh), c0);
    c1 = wmma_f16(xa[kc], load_bfrag(w1t, 128, 16, kc * 32, m, hh), c1);
  }
  bias_lrelu_to_lds(c0, c1, b1, lds, m, hh);
  v16h a2 = load_afrag_lds(lds, m, hh);
  v8f d0 = {}, d1 = {};
  d0 = wmma_f16(a2, load_bfrag(w2t, 32, 0,  0, m, hh), d0);
  d1 = wmma_f16(a2, load_bfrag(w2t, 32, 16, 0, m, hh), d1);
  bias_lrelu_to_lds(d0, d1, b2, lds, m, hh);
  v16h a3 = load_afrag_lds(lds, m, hh);
  v8f e0 = {}, e1 = {};
  e0 = wmma_f16(a3, load_bfrag(w3t, 32, 0,  0, m, hh), e0);
  e1 = wmma_f16(a3, load_bfrag(w3t, 32, 16, 0, m, hh), e1);
  float bb0 = b3[m], bb1 = b3[m + 16];
#pragma unroll
  for (int r = 0; r < 8; ++r) { e0[r] += bb0; e1[r] += bb1; }
  o0 = e0; o1 = e1;
}

// ------------------------------------------------------------- prep kernels
__global__ void __launch_bounds__(256)
k_init(int* sA, int* sB, int* sC, int* sD,
       int* binsA, int* fillA, int* binsB, int* fillB,
       int* binsC, int* fillC, int* binsD, int* fillD,
       float* ceAcc, float* mkAcc) {
  int i = blockIdx.x * blockDim.x + threadIdx.x;
  if (i < NPIX + 256)  { sA[i] = -1; sB[i] = -1; }
  if (i < NPIX + 4096) { sC[i] = -1; sD[i] = -1; }
  if (i < 16)  { binsA[i] = 0; fillA[i] = 0; binsB[i] = 0; fillB[i] = 0; }
  if (i < 256) { binsC[i] = 0; fillC[i] = 0; binsD[i] = 0; fillD[i] = 0; }
  if (i < 7) ceAcc[i] = 0.f;
  if (i < 6) mkAcc[i] = 0.f;
}

// NCHW fp32 -> [N,128] f16 (reads coalesced over pixels; 16B packed stores).
__global__ void __launch_bounds__(256)
k_xpose(const float* __restrict__ x, _Float16* __restrict__ xT) {
  int n = blockIdx.x * blockDim.x + threadIdx.x;
  if (n >= NPIX) return;
#pragma unroll 4
  for (int c0 = 0; c0 < 128; c0 += 8) {
    v8h pk;
#pragma unroll
    for (int j = 0; j < 8; ++j) pk[j] = (_Float16)x[(size_t)(c0 + j) * NPIX + n];
    *(v8h*)(xT + (size_t)n * 128 + c0) = pk;
  }
}

__global__ void __launch_bounds__(256)
k_cvt(const float* __restrict__ s, _Float16* __restrict__ d, int n) {
  int i = blockIdx.x * blockDim.x + threadIdx.x;
  if (i < n) d[i] = (_Float16)s[i];
}

// [E][K][32] fp32 -> [E][32][K] f16 (K contiguous for B fragments).
__global__ void __launch_bounds__(256)
k_wxpose(const float* __restrict__ src, _Float16* __restrict__ dst, int K, int total) {
  int i = blockIdx.x * blockDim.x + threadIdx.x;
  if (i >= total) return;
  int n = i & 31;
  int k = (i >> 5) % K;
  int e = i / (K * 32);
  dst[((size_t)e * 32 + n) * K + k] = (_Float16)src[i];
}

// ---------------------------------------------------- bucketing (MoE group)
DEVINL int bin_of(int v, int mode) {
  if (mode == 0) return v;                 // pre-computed expert index
  int ig = clampi(v, 0, 4095);             // from inds_gt
  return mode == 1 ? (ig >> 8) : (ig >> 4);
}

__global__ void __launch_bounds__(256)
k_hist(const int* __restrict__ src, int mode, int* __restrict__ bins) {
  int i = blockIdx.x * blockDim.x + threadIdx.x;
  if (i >= NPIX) return;
  atomicAdd(bins + bin_of(src[i], mode), 1);
}

__global__ void k_scan(const int* __restrict__ bins, int* __restrict__ tileOff, int nb) {
  if (threadIdx.x == 0 && blockIdx.x == 0) {
    int acc = 0;
    for (int e = 0; e < nb; ++e) { tileOff[e] = acc; acc += (bins[e] + 15) >> 4; }
    tileOff[nb] = acc;   // total tiles
  }
}

__global__ void __launch_bounds__(256)
k_scatter(const int* __restrict__ src, int mode, const int* __restrict__ tileOff,
          int* __restrict__ fill, int* __restrict__ sorted) {
  int i = blockIdx.x * blockDim.x + threadIdx.x;
  if (i >= NPIX) return;
  int b = bin_of(src[i], mode);
  int pos = tileOff[b] * 16 + atomicAdd(fill + b, 1);
  sorted[pos] = i;
}

DEVINL int find_expert(const int* tileOff, int nb, int t) {
  int lo = 0, hi = nb - 1;
  while (lo < hi) {
    int mid = (lo + hi + 1) >> 1;
    if (tileOff[mid] <= t) lo = mid; else hi = mid - 1;
  }
  return lo;
}

// -------------------------------------------------------------- coarse MLP
__global__ void __launch_bounds__(256)
k_coarse(const _Float16* __restrict__ xT,
         const _Float16* __restrict__ w1t, const float* __restrict__ b1,
         const _Float16* __restrict__ w2t, const float* __restrict__ b2,
         const _Float16* __restrict__ w3t, const float* __restrict__ b3,
         const int* __restrict__ inds_gt,
         int* __restrict__ inds1, float* __restrict__ ceAcc) {
  __shared__ __align__(16) _Float16 lt[8][16 * 32];
  int lane = threadIdx.x & 31, m = lane & 15, hh = lane >> 4;
  int wv = threadIdx.x >> 5;
  int pixBase = blockIdx.x * 128 + wv * 16;
  const _Float16* xrow = xT + (size_t)(pixBase + m) * 128;
  v16h xa[4];
#pragma unroll
  for (int kc = 0; kc < 4; ++kc) xa[kc] = load_afrag_row(xrow, kc * 32, hh);
  _Float16* lds = lt[wv];

  v8f c0 = {}, c1 = {};
#pragma unroll
  for (int kc = 0; kc < 4; ++kc) {
    c0 = wmma_f16(xa[kc], load_bfrag(w1t, 128, 0,  kc * 32, m, hh), c0);
    c1 = wmma_f16(xa[kc], load_bfrag(w1t, 128, 16, kc * 32, m, hh), c1);
  }
  bias_lrelu_to_lds(c0, c1, b1, lds, m, hh);
  v16h a2 = load_afrag_lds(lds, m, hh);
  v8f d0 = {}, d1 = {};
  d0 = wmma_f16(a2, load_bfrag(w2t, 32, 0,  0, m, hh), d0);
  d1 = wmma_f16(a2, load_bfrag(w2t, 32, 16, 0, m, hh), d1);
  bias_lrelu_to_lds(d0, d1, b2, lds, m, hh);
  v16h a3 = load_afrag_lds(lds, m, hh);
  v8f e0 = {};
  e0 = wmma_f16(a3, load_bfrag(w3t, 32, 0, 0, m, hh), e0);  // 16 logits

  float bb = b3[m];
  int igm = clampi(inds_gt[pixBase + m], 0, 4095);
  float lce = 0.f;
#pragma unroll
  for (int r = 0; r < 8; ++r) {
    int row = r + 8 * hh;
    float v = e0[r] + bb;
    float V = v; int I = m;
    rargmax16(V, I);                       // V = rowmax, I = argmax
    if (m == 0) inds1[pixBase + row] = I;
    int igr = __shfl(igm, row, 32);
    int tt  = igr >> 8;                    // i1_gt
    float se = rsum16(expf(v - V));
    float tv = __shfl(v, (lane & 16) | tt, 32);
    lce -= (tv - V) - logf(se);
  }
  float tot = __shfl(lce, 0, 32) + __shfl(lce, 16, 32);
  if (lane == 0) atomicAdd(ceAcc + 0, tot);
}

// --------------------------------------------------- grouped expert kernels
__global__ void __launch_bounds__(32)
k_group_route(const _Float16* __restrict__ xT,
              const int* __restrict__ sorted, const int* __restrict__ tileOff, int nb,
              const _Float16* __restrict__ w1t, const float* __restrict__ b1,
              const _Float16* __restrict__ w2t, const float* __restrict__ b2,
              const _Float16* __restrict__ w3t, const float* __restrict__ b3,
              int stage, int* __restrict__ outIdx, float* __restrict__ outF) {
  __shared__ __align__(16) _Float16 lt[16 * 32];
  int t = blockIdx.x;
  if (t >= tileOff[nb]) return;            // uniform per block
  int lane = threadIdx.x & 31, m = lane & 15, hh = lane >> 4;
  int e = find_expert(tileOff, nb, t);
  // expert id is data-dependent: kick the weight block toward the caches
  // before the A-fragment gathers (emits global_prefetch_b8).
  __builtin_prefetch(w1t + (size_t)e * 32 * 128 + lane * 256, 0, 1);
  int base = t * 16;
  int pix = sorted[base + m];
  const _Float16* xrow = pix >= 0 ? xT + (size_t)pix * 128 : nullptr;
  v16h xa[4];
#pragma unroll
  for (int kc = 0; kc < 4; ++kc) xa[kc] = load_afrag_row(xrow, kc * 32, hh);
  v8f o0, o1;
  mlp_128_32_32_32(xa, w1t + (size_t)e * 32 * 128, b1 + e * 32,
                       w2t + (size_t)e * 32 * 32,  b2 + e * 32,
                       w3t + (size_t)e * 32 * 32,  b3 + e * 32,
                   lt, m, hh, o0, o1);
  int lim = (stage == 2) ? 255 : 4095;
#pragma unroll
  for (int r = 0; r < 8; ++r) {
    float v0 = o0[r], v1 = o1[r];
    float V; int I;
    if (v1 > v0) { V = v1; I = m + 16; } else { V = v0; I = m; }
    rargmax16(V, I);
    if (m == 0) {
      int p = sorted[base + r + 8 * hh];
      if (p >= 0) {
        int vi = clampi(e * 16 + I - 8, 0, lim);
        if (stage == 2) outIdx[p] = vi;
        else            outF[p]  = (float)vi;   // inds123 as float output
      }
    }
  }
}

__global__ void __launch_bounds__(32)
k_group_loss(const _Float16* __restrict__ xT,
             const int* __restrict__ sorted, const int* __restrict__ tileOff, int nb,
             const _Float16* __restrict__ w1t, const float* __restrict__ b1,
             const _Float16* __restrict__ w2t, const float* __restrict__ b2,
             const _Float16* __restrict__ w3t, const float* __restrict__ b3,
             int stage, const int* __restrict__ inds_gt,
             float* __restrict__ ceAcc, float* __restrict__ mkAcc,
             int ceBase, int mkBase) {
  // one private transpose tile per GT-offset so the three expert MLPs are not
  // serialized on the same LDS bytes; the unrolled loop lets the scheduler
  // interleave their 36 WMMAs and weight loads (they share the x A-fragments).
  __shared__ __align__(16) _Float16 lt[3][16 * 32];
  int t = blockIdx.x;
  if (t >= tileOff[nb]) return;
  int lane = threadIdx.x & 31, m = lane & 15, hh = lane >> 4;
  int e = find_expert(tileOff, nb, t);
  __builtin_prefetch(w1t + (size_t)e * 32 * 128 + lane * 256, 0, 1);
  int base = t * 16;
  int pix = sorted[base + m];
  int igm = pix >= 0 ? clampi(inds_gt[pix], 0, 4095) : 0;
  const _Float16* xrow = pix >= 0 ? xT + (size_t)pix * 128 : nullptr;
  v16h xa[4];                              // A fragments shared by all 3 offsets
#pragma unroll
  for (int kc = 0; kc < 4; ++kc) xa[kc] = load_afrag_row(xrow, kc * 32, hh);

#pragma unroll
  for (int o = 0; o < 3; ++o) {
    int eo = clampi(e + o - 1, 0, nb - 1);
    v8f o0, o1;
    mlp_128_32_32_32(xa, w1t + (size_t)eo * 32 * 128, b1 + eo * 32,
                         w2t + (size_t)eo * 32 * 32,  b2 + eo * 32,
                         w3t + (size_t)eo * 32 * 32,  b3 + eo * 32,
                     lt[o], m, hh, o0, o1);
    float lce = 0.f, lmk = 0.f;
#pragma unroll
    for (int r = 0; r < 8; ++r) {
      int row  = r + 8 * hh;
      int prow = __shfl(pix, row, 32);
      int igr  = __shfl(igm, row, 32);
      int iT, mk;
      if (stage == 2) { int i2 = (igr >> 4) - eo * 16 + 8; mk = (i2 >= 0 && i2 < 32); iT = clampi(i2, 0, 31); }
      else            { int i3 = igr        - eo * 16 + 8; mk = (i3 >= 0 && i3 < 32); iT = clampi(i3, 0, 31); }
      float v0 = o0[r], v1 = o1[r];
      float mx = rmax16(fmaxf(v0, v1));
      float se = rsum16(expf(v0 - mx) + expf(v1 - mx));
      float sel = (iT >= 16) ? v1 : v0;    // iT uniform within half-group
      float tv  = __shfl(sel, (lane & 16) | (iT & 15), 32);
      if (prow >= 0) { lce -= (tv - mx) - logf(se); lmk += (float)mk; }
    }
    float ce2 = __shfl(lce, 0, 32) + __shfl(lce, 16, 32);
    float mk2 = __shfl(lmk, 0, 32) + __shfl(lmk, 16, 32);
    if (lane == 0) { atomicAdd(ceAcc + ceBase + o, ce2); atomicAdd(mkAcc + mkBase + o, mk2); }
  }
}

__global__ void k_finalize(const float* __restrict__ ce, const float* __restrict__ mk,
                           float* __restrict__ out) {
  if (threadIdx.x == 0 && blockIdx.x == 0) {
    const float inv = 1.0f / (float)NPIX;
    out[NPIX + 0] = ce[0] * inv;
#pragma unroll
    for (int i = 0; i < 3; ++i) out[NPIX + 1 + i] = (ce[1 + i] * inv) * (mk[i] * inv);
#pragma unroll
    for (int i = 0; i < 3; ++i) out[NPIX + 4 + i] = (ce[4 + i] * inv) * (mk[3 + i] * inv);
  }
}

// ------------------------------------------------------------------- launch
extern "C" void kernel_launch(void* const* d_in, const int* in_sizes, int n_in,
                              void* d_out, int out_size, void* d_ws, size_t ws_size,
                              hipStream_t stream) {
  (void)in_sizes; (void)n_in; (void)out_size; (void)ws_size;
  const float* x_in    = (const float*)d_in[0];
  const int*   inds_gt = (const int*)d_in[1];
  const float* conv_w1 = (const float*)d_in[2];
  const float* conv_b1 = (const float*)d_in[3];
  const float* conv_w2 = (const float*)d_in[4];
  const float* conv_b2 = (const float*)d_in[5];
  const float* conv_w3 = (const float*)d_in[6];
  const float* conv_b3 = (const float*)d_in[7];
  const float* cm1_w1 = (const float*)d_in[8];   const float* cm1_b1 = (const float*)d_in[9];
  const float* cm1_w2 = (const float*)d_in[10];  const float* cm1_b2 = (const float*)d_in[11];
  const float* cm1_w3 = (const float*)d_in[12];  const float* cm1_b3 = (const float*)d_in[13];
  const float* cm2_w1 = (const float*)d_in[14];  const float* cm2_b1 = (const float*)d_in[15];
  const float* cm2_w2 = (const float*)d_in[16];  const float* cm2_b2 = (const float*)d_in[17];
  const float* cm2_w3 = (const float*)d_in[18];  const float* cm2_b3 = (const float*)d_in[19];
  float* out = (float*)d_out;

  // workspace arena (~14.6 MB total)
  char* p = (char*)d_ws;
  auto alloc = [&](size_t bytes) -> char* {
    char* r = p; p += (bytes + 255) & ~(size_t)255; return r;
  };
  _Float16* xT    = (_Float16*)alloc((size_t)NPIX * 128 * 2);
  _Float16* cw1t  = (_Float16*)alloc(32 * 128 * 2);
  _Float16* cw2t  = (_Float16*)alloc(32 * 32 * 2);
  _Float16* cw3t  = (_Float16*)alloc(16 * 32 * 2);
  _Float16* m1w1t = (_Float16*)alloc((size_t)16 * 32 * 128 * 2);
  _Float16* m1w2t = (_Float16*)alloc((size_t)16 * 32 * 32 * 2);
  _Float16* m1w3t = (_Float16*)alloc((size_t)16 * 32 * 32 * 2);
  _Float16* m2w1t = (_Float16*)alloc((size_t)256 * 32 * 128 * 2);
  _Float16* m2w2t = (_Float16*)alloc((size_t)256 * 32 * 32 * 2);
  _Float16* m2w3t = (_Float16*)alloc((size_t)256 * 32 * 32 * 2);
  int* inds1  = (int*)alloc(NPIX * 4);
  int* inds12 = (int*)alloc(NPIX * 4);
  int* binsA = (int*)alloc(16 * 4);  int* offA = (int*)alloc(17 * 4);  int* fillA = (int*)alloc(16 * 4);
  int* sortA = (int*)alloc((NPIX + 256) * 4);
  int* binsB = (int*)alloc(16 * 4);  int* offB = (int*)alloc(17 * 4);  int* fillB = (int*)alloc(16 * 4);
  int* sortB = (int*)alloc((NPIX + 256) * 4);
  int* binsC = (int*)alloc(256 * 4); int* offC = (int*)alloc(257 * 4); int* fillC = (int*)alloc(256 * 4);
  int* sortC = (int*)alloc((NPIX + 4096) * 4);
  int* binsD = (int*)alloc(256 * 4); int* offD = (int*)alloc(257 * 4); int* fillD = (int*)alloc(256 * 4);
  int* sortD = (int*)alloc((NPIX + 4096) * 4);
  float* ceAcc = (float*)alloc(7 * 4);
  float* mkAcc = (float*)alloc(6 * 4);

  const dim3 B256(256), B32(32);
  const dim3 gPix((NPIX + 255) / 256);
  const int tiles16  = NPIX / 16 + 16;    // padded-bucket worst case (16 bins)
  const int tiles256 = NPIX / 16 + 256;   // (256 bins)

  k_init<<<dim3((NPIX + 4096 + 255) / 256), B256, 0, stream>>>(
      sortA, sortB, sortC, sortD, binsA, fillA, binsB, fillB,
      binsC, fillC, binsD, fillD, ceAcc, mkAcc);

  // data/weight prep
  k_xpose<<<gPix, B256, 0, stream>>>(x_in, xT);
  k_cvt<<<dim3(16), B256, 0, stream>>>(conv_w1, cw1t, 32 * 128);
  k_cvt<<<dim3(4),  B256, 0, stream>>>(conv_w2, cw2t, 32 * 32);
  k_cvt<<<dim3(2),  B256, 0, stream>>>(conv_w3, cw3t, 16 * 32);
  k_wxpose<<<dim3(256),  B256, 0, stream>>>(cm1_w1, m1w1t, 128, 16 * 128 * 32);
  k_wxpose<<<dim3(64),   B256, 0, stream>>>(cm1_w2, m1w2t, 32,  16 * 32 * 32);
  k_wxpose<<<dim3(64),   B256, 0, stream>>>(cm1_w3, m1w3t, 32,  16 * 32 * 32);
  k_wxpose<<<dim3(4096), B256, 0, stream>>>(cm2_w1, m2w1t, 128, 256 * 128 * 32);
  k_wxpose<<<dim3(1024), B256, 0, stream>>>(cm2_w2, m2w2t, 32,  256 * 32 * 32);
  k_wxpose<<<dim3(1024), B256, 0, stream>>>(cm2_w3, m2w3t, 32,  256 * 32 * 32);

  // coarse stage + loss0 + inds1
  k_coarse<<<dim3(NPIX / 128), B256, 0, stream>>>(
      xT, cw1t, conv_b1, cw2t, conv_b2, cw3t, conv_b3, inds_gt, inds1, ceAcc);

  // routing: stage2 (experts = inds1) -> inds12
  k_hist<<<gPix, B256, 0, stream>>>(inds1, 0, binsA);
  k_scan<<<dim3(1), B32, 0, stream>>>(binsA, offA, 16);
  k_scatter<<<gPix, B256, 0, stream>>>(inds1, 0, offA, fillA, sortA);
  k_group_route<<<dim3(tiles16), B32, 0, stream>>>(
      xT, sortA, offA, 16, m1w1t, cm1_b1, m1w2t, cm1_b2, m1w3t, cm1_b3,
      2, inds12, nullptr);

  // routing: stage3 (experts = inds12) -> inds123 into d_out
  k_hist<<<gPix, B256, 0, stream>>>(inds12, 0, binsC);
  k_scan<<<dim3(1), B32, 0, stream>>>(binsC, offC, 256);
  k_scatter<<<gPix, B256, 0, stream>>>(inds12, 0, offC, fillC, sortC);
  k_group_route<<<dim3(tiles256), B32, 0, stream>>>(
      xT, sortC, offC, 256, m2w1t, cm2_b1, m2w2t, cm2_b2, m2w3t, cm2_b3,
      3, nullptr, out);

  // losses 1..3: stage2 at i1_gt-1/0/+1 (bucket by i1_gt = inds_gt>>8)
  k_hist<<<gPix, B256, 0, stream>>>(inds_gt, 1, binsB);
  k_scan<<<dim3(1), B32, 0, stream>>>(binsB, offB, 16);
  k_scatter<<<gPix, B256, 0, stream>>>(inds_gt, 1, offB, fillB, sortB);
  k_group_loss<<<dim3(tiles16), B32, 0, stream>>>(
      xT, sortB, offB, 16, m1w1t, cm1_b1, m1w2t, cm1_b2, m1w3t, cm1_b3,
      2, inds_gt, ceAcc, mkAcc, 1, 0);

  // losses 4..6: stage3 at i12_gt-1/0/+1 (bucket by i12_gt = inds_gt>>4)
  k_hist<<<gPix, B256, 0, stream>>>(inds_gt, 2, binsD);
  k_scan<<<dim3(1), B32, 0, stream>>>(binsD, offD, 256);
  k_scatter<<<gPix, B256, 0, stream>>>(inds_gt, 2, offD, fillD, sortD);
  k_group_loss<<<dim3(tiles256), B32, 0, stream>>>(
      xT, sortD, offD, 256, m2w1t, cm2_b1, m2w2t, cm2_b2, m2w3t, cm2_b3,
      3, inds_gt, ceAcc, mkAcc, 4, 3);

  k_finalize<<<dim3(1), B32, 0, stream>>>(ceAcc, mkAcc, out);
}
